// RoFormerAttention_63264868270200
// MI455X (gfx1250) — matrix operand order repaired and checked by reading, not compiled, not run on hardware
//
#include <hip/hip_runtime.h>
#include <cmath>

#define DEV static __device__ __forceinline__

constexpr int Bz  = 2;
constexpr int Ssz = 2048;
constexpr int Dm  = 1024;
constexpr int Hn  = 16;
constexpr int HDm = 64;
constexpr int BS  = Bz * Ssz;   // 4096

typedef __attribute__((ext_vector_type(16))) __bf16 v16bf;
typedef __attribute__((ext_vector_type(8)))  float  v8f;

union Fbf { v16bf v; uint4 q[2]; };

DEV unsigned short f2bf(float f) {
    unsigned u = __float_as_uint(f);
    unsigned r = u + 0x7FFFu + ((u >> 16) & 1u);   // round-to-nearest-even
    return (unsigned short)(r >> 16);
}

DEV v8f wmma_bf16(v16bf a, v16bf b, v8f c) {
    return __builtin_amdgcn_wmma_f32_16x16x32_bf16(
        /*neg_a=*/false, a, /*neg_b=*/false, b,
        /*c_mod=*/(short)0, c, /*reuse_a=*/false, /*reuse_b=*/false);
}

// A fragment: 16(M) x 32(K) tile from row-major bf16, row stride lda.
// Layout (ISA 7.12.2): lanes 0-15 -> M=lane, K={0..7,16..23}; lanes 16-31 -> K={8..15,24..31}
DEV v16bf load_a(const unsigned short* __restrict__ src, int lda, int m0, int k0, int lane) {
    int m  = m0 + (lane & 15);
    int kb = k0 + ((lane >> 4) << 3);
    Fbf f;
    f.q[0] = *reinterpret_cast<const uint4*>(src + (size_t)m * lda + kb);
    f.q[1] = *reinterpret_cast<const uint4*>(src + (size_t)m * lda + kb + 16);
    return f.v;
}

// B fragment: 32(K) x 16(N) where B[k,n] = rows[n][k] (rows row-major, stride ldw).
// Works on global or LDS pointers.
DEV v16bf load_b(const unsigned short* rows, int ldw, int n0, int k0, int lane) {
    int n  = n0 + (lane & 15);
    int kb = k0 + ((lane >> 4) << 3);
    const uint4* p = reinterpret_cast<const uint4*>(rows + (size_t)n * ldw + kb);
    Fbf f;
    f.q[0] = p[0];
    f.q[1] = p[2];
    return f.v;
}

// ---- CDNA5 async LDS DMA helpers ------------------------------------------
DEV void async_b128(unsigned lds_off, const unsigned short* gaddr) {
    asm volatile("global_load_async_to_lds_b128 %0, %1, off"
                 :: "v"(lds_off), "v"((unsigned long long)(uintptr_t)gaddr)
                 : "memory");
}
DEV void wait_async_le2() { asm volatile("s_wait_asynccnt 0x2" ::: "memory"); }
DEV void wait_async_0()   { asm volatile("s_wait_asynccnt 0x0" ::: "memory"); }
DEV unsigned lds_off(const void* p) { return (unsigned)(uintptr_t)p; }

// ---------------------------------------------------------------- conversion
__global__ void conv_bf16_kernel(const float* __restrict__ src,
                                 unsigned short* __restrict__ dst, int n) {
    int i = blockIdx.x * blockDim.x + threadIdx.x;
    int stride = gridDim.x * blockDim.x;
    for (; i < n; i += stride) dst[i] = f2bf(src[i]);
}

// ------------------------------------------------- QKV projection + rotary
// one wave = 32(M) x 64(N) tile: two A frags share every B frag (2x B reuse)
__global__ void qkv_kernel(const unsigned short* __restrict__ Xb,
                           const unsigned short* __restrict__ Wq,
                           const unsigned short* __restrict__ Wk,
                           const unsigned short* __restrict__ Wv,
                           const float* __restrict__ sinp,   // [S, HD/2]
                           const float* __restrict__ cosp,   // [S, HD/2]
                           unsigned short* __restrict__ Qb,  // [B,H,S,HD]
                           unsigned short* __restrict__ Kb,  // [B,H,S,HD]
                           unsigned short* __restrict__ Vt)  // [B,H,HD,S]
{
    int lane = threadIdx.x & 31;
    int wid  = (blockIdx.x * blockDim.x + threadIdx.x) >> 5;
    int mat  = wid / (128 * 16);          // 0=q,1=k,2=v
    int rem  = wid % (128 * 16);
    int mt   = rem >> 4;                  // 0..127 (row tiles of 32 over BS)
    int ht   = rem & 15;                  // head index
    const unsigned short* W = (mat == 0) ? Wq : (mat == 1) ? Wk : Wv;
    int m0 = mt * 32;
    int n0 = ht * 64;

    v8f acc[2][4] = {};
    for (int k0 = 0; k0 < Dm; k0 += 32) {
        v16bf a0 = load_a(Xb, Dm, m0,      k0, lane);
        v16bf a1 = load_a(Xb, Dm, m0 + 16, k0, lane);
#pragma unroll
        for (int c = 0; c < 4; ++c) {
            v16bf b = load_b(W, Dm, n0 + c * 16, k0, lane);
            acc[0][c] = wmma_bf16(a0, b, acc[0][c]);
            acc[1][c] = wmma_bf16(a1, b, acc[1][c]);
        }
    }

    int half = lane >> 4;
    int nloc = lane & 15;
    int b = m0 / Ssz;                     // 32-row tile never crosses a batch
#pragma unroll
    for (int r = 0; r < 2; ++r) {
        int mr = m0 + r * 16;
        // Rotary: pair (2i,2i+1) = adjacent columns = adjacent lanes in C layout.
#pragma unroll
        for (int c = 0; c < 4; ++c) {
            int col  = c * 16 + nloc;
            int pidx = col >> 1;
            bool even = (col & 1) == 0;
#pragma unroll
            for (int i = 0; i < 8; ++i) {
                int s = (mr + i + half * 8) & (Ssz - 1);
                float cs = cosp[s * (HDm / 2) + pidx];
                float sn = sinp[s * (HDm / 2) + pidx];
                float own   = acc[r][c][i];
                float other = __shfl_xor(own, 1, 32);
                acc[r][c][i] = even ? (own * cs - other * sn)
                                    : (own * cs + other * sn);
            }
        }
#pragma unroll
        for (int c = 0; c < 4; ++c) {
            int col = c * 16 + nloc;
#pragma unroll
            for (int i = 0; i < 8; ++i) {
                int s = (mr + i + half * 8) & (Ssz - 1);
                unsigned short v = f2bf(acc[r][c][i]);
                if (mat == 0)
                    Qb[(((size_t)(b * Hn + ht)) * Ssz + s) * HDm + col] = v;
                else if (mat == 1)
                    Kb[(((size_t)(b * Hn + ht)) * Ssz + s) * HDm + col] = v;
                else
                    Vt[(((size_t)(b * Hn + ht)) * HDm + col) * Ssz + s] = v;
            }
        }
    }
}

// ----------------------------------------------------- flash attention core
// block = 8 waves sharing one (b,h); 128 queries/block; K/V tiles staged in
// LDS via async DMA (double buffered), 8x reduction in K/V read traffic.
__global__ void attn_kernel(const unsigned short* __restrict__ Qb,
                            const unsigned short* __restrict__ Kb,
                            const unsigned short* __restrict__ Vt,
                            unsigned short* __restrict__ Ctx)  // [BS, D] bf16
{
    __shared__ __align__(16) unsigned short KtS[2][32 * 64];  // 2 x 4KB
    __shared__ __align__(16) unsigned short VtS[2][64 * 32];  // 2 x 4KB
    __shared__ __align__(16) unsigned short pbuf[8][16 * 32]; // 8 x 1KB

    int tid  = threadIdx.x;
    int lane = tid & 31;
    int w    = tid >> 5;
    int bh   = blockIdx.x >> 4;           // b*H + h
    int qb   = blockIdx.x & 15;           // query block (128 rows each)
    const unsigned short* Q = Qb + (size_t)bh * Ssz * HDm;
    const unsigned short* K = Kb + (size_t)bh * Ssz * HDm;
    const unsigned short* V = Vt + (size_t)bh * HDm * Ssz;
    int m0 = qb * 128 + w * 16;

    // per-thread DMA chunk assignment (one b128 per thread per tile)
    int kr = tid >> 3, kcol = (tid & 7) * 8;   // K tile: 32 rows x 64 elems
    int vr = tid >> 2, vcol = (tid & 3) * 8;   // V tile: 64 rows x 32 elems

    v16bf qa0 = load_a(Q, HDm, m0, 0,  lane);
    v16bf qa1 = load_a(Q, HDm, m0, 32, lane);

    v8f O[4] = {};
    float mrun[8], lrun[8];
#pragma unroll
    for (int i = 0; i < 8; ++i) { mrun[i] = -1e30f; lrun[i] = 0.f; }
    const float scale = 0.125f;           // 1/sqrt(64)
    int half = lane >> 4, nloc = lane & 15;

    // prologue: issue tile 0
    async_b128(lds_off(&KtS[0][kr * 64 + kcol]), K + (size_t)kr * HDm + kcol);
    async_b128(lds_off(&VtS[0][vr * 32 + vcol]), V + (size_t)vr * Ssz + vcol);
    int p = 0;

    for (int kc = 0; kc < Ssz; kc += 32) {
        bool has_next = (kc + 32) < Ssz;
        if (has_next) {
            int kn = kc + 32;
            async_b128(lds_off(&KtS[1 - p][kr * 64 + kcol]),
                       K + (size_t)(kn + kr) * HDm + kcol);
            async_b128(lds_off(&VtS[1 - p][vr * 32 + vcol]),
                       V + (size_t)vr * Ssz + kn + vcol);
            wait_async_le2();             // oldest tile (current) has landed
        } else {
            wait_async_0();
        }
        __syncthreads();                  // publish LDS tile to all waves

        const unsigned short* Kt = KtS[p];
        const unsigned short* Vl = VtS[p];

        // scores 16x32 as two 16x16 tiles (K-dim = HD = 64 -> 2 wmma each)
        v8f s0 = {}, s1 = {};
        {
            v16bf b00 = load_b(Kt, HDm, 0,  0,  lane);
            v16bf b01 = load_b(Kt, HDm, 0,  32, lane);
            s0 = wmma_bf16(qa0, b00, s0);
            s0 = wmma_bf16(qa1, b01, s0);
            v16bf b10 = load_b(Kt, HDm, 16, 0,  lane);
            v16bf b11 = load_b(Kt, HDm, 16, 32, lane);
            s1 = wmma_bf16(qa0, b10, s1);
            s1 = wmma_bf16(qa1, b11, s1);
        }
        // online softmax per row (rows live across 16-lane halves)
#pragma unroll
        for (int i = 0; i < 8; ++i) {
            float a0 = s0[i] * scale, a1 = s1[i] * scale;
            float mx = fmaxf(a0, a1);
#pragma unroll
            for (int off = 1; off < 16; off <<= 1)
                mx = fmaxf(mx, __shfl_xor(mx, off, 32));
            float mnew = fmaxf(mrun[i], mx);
            float corr = __expf(mrun[i] - mnew);
            float p0 = __expf(a0 - mnew);
            float p1 = __expf(a1 - mnew);
            float r = p0 + p1;
#pragma unroll
            for (int off = 1; off < 16; off <<= 1)
                r += __shfl_xor(r, off, 32);
            lrun[i] = lrun[i] * corr + r;
            mrun[i] = mnew;
#pragma unroll
            for (int c = 0; c < 4; ++c) O[c][i] *= corr;
            int row = i + half * 8;
            pbuf[w][row * 32 + nloc]      = f2bf(p0);
            pbuf[w][row * 32 + 16 + nloc] = f2bf(p1);
        }
        // restripe P (C layout) -> A fragment via per-wave LDS
        v16bf pa;
        {
            int m  = lane & 15;
            int kb = half * 8;
            Fbf f;
            f.q[0] = *reinterpret_cast<const uint4*>(&pbuf[w][m * 32 + kb]);
            f.q[1] = *reinterpret_cast<const uint4*>(&pbuf[w][m * 32 + kb + 16]);
            pa = f.v;
        }
        // ctx += P(16x32) @ V(32x64); V tile is transposed: rows = hd
#pragma unroll
        for (int c = 0; c < 4; ++c) {
            v16bf vb = load_b(Vl, 32, c * 16, 0, lane);
            O[c] = wmma_bf16(pa, vb, O[c]);
        }
        __syncthreads();                  // all waves done reading buf[p]
        p ^= 1;
    }

    int b = bh >> 4, h = bh & 15;
#pragma unroll
    for (int i = 0; i < 8; ++i) {
        float inv = 1.f / lrun[i];
        int s = m0 + i + half * 8;
#pragma unroll
        for (int c = 0; c < 4; ++c) {
            Ctx[((size_t)(b * Ssz + s)) * Dm + h * HDm + c * 16 + nloc] =
                f2bf(O[c][i] * inv);
        }
    }
}

// -------------------------------------- output projection + bias + residual
__global__ void oproj_kernel(const unsigned short* __restrict__ Ctx,
                             const unsigned short* __restrict__ Wo,
                             const float* __restrict__ bo,
                             const float* __restrict__ X,
                             float* __restrict__ Y)
{
    int lane = threadIdx.x & 31;
    int wid  = (blockIdx.x * blockDim.x + threadIdx.x) >> 5;
    int mt = wid >> 4;                    // 128 row tiles of 32
    int nt = wid & 15;                    // 16 col tiles of 64
    int m0 = mt * 32, n0 = nt * 64;

    v8f acc[2][4] = {};
    for (int k0 = 0; k0 < Dm; k0 += 32) {
        v16bf a0 = load_a(Ctx, Dm, m0,      k0, lane);
        v16bf a1 = load_a(Ctx, Dm, m0 + 16, k0, lane);
#pragma unroll
        for (int c = 0; c < 4; ++c) {
            v16bf b = load_b(Wo, Dm, n0 + c * 16, k0, lane);
            acc[0][c] = wmma_bf16(a0, b, acc[0][c]);
            acc[1][c] = wmma_bf16(a1, b, acc[1][c]);
        }
    }
    int half = lane >> 4, nloc = lane & 15;
#pragma unroll
    for (int r = 0; r < 2; ++r) {
#pragma unroll
        for (int c = 0; c < 4; ++c) {
            int n = n0 + c * 16 + nloc;
            float bias = bo[n];
#pragma unroll
            for (int i = 0; i < 8; ++i) {
                int m = m0 + r * 16 + i + half * 8;
                Y[(size_t)m * Dm + n] = acc[r][c][i] + bias + X[(size_t)m * Dm + n];
            }
        }
    }
}

// ----------------------------------------------------------- RMS-style norm
__global__ void norm_kernel(const float* __restrict__ Y,
                            const float* __restrict__ scale,
                            float* __restrict__ out)
{
    int lane = threadIdx.x & 31;
    int row  = (blockIdx.x * blockDim.x + threadIdx.x) >> 5;
    const float* y = Y + (size_t)row * Dm;
    float4 vals[8];
    float ss = 0.f;
#pragma unroll
    for (int j = 0; j < 8; ++j) {
        vals[j] = reinterpret_cast<const float4*>(y)[lane + j * 32];
        ss += vals[j].x * vals[j].x + vals[j].y * vals[j].y +
              vals[j].z * vals[j].z + vals[j].w * vals[j].w;
    }
#pragma unroll
    for (int off = 1; off < 32; off <<= 1) ss += __shfl_xor(ss, off, 32);
    float rms = sqrtf(ss * (1.0f / Dm));
    float inv = 1.f / (rms + 1e-8f);
    float* o = out + (size_t)row * Dm;
#pragma unroll
    for (int j = 0; j < 8; ++j) {
        float4 sc = reinterpret_cast<const float4*>(scale)[lane + j * 32];
        float4 r;
        r.x = vals[j].x * inv * sc.x;
        r.y = vals[j].y * inv * sc.y;
        r.z = vals[j].z * inv * sc.z;
        r.w = vals[j].w * inv * sc.w;
        reinterpret_cast<float4*>(o)[lane + j * 32] = r;
    }
}

extern "C" void kernel_launch(void* const* d_in, const int* in_sizes, int n_in,
                              void* d_out, int out_size, void* d_ws, size_t ws_size,
                              hipStream_t stream) {
    const float* X     = (const float*)d_in[0];
    const float* sinp  = (const float*)d_in[1];   // [1,1,S,HD/2]
    const float* cosp  = (const float*)d_in[2];
    const float* wq    = (const float*)d_in[3];
    const float* wk    = (const float*)d_in[4];
    const float* wv    = (const float*)d_in[5];
    const float* wo    = (const float*)d_in[6];
    const float* bo    = (const float*)d_in[7];
    const float* scale = (const float*)d_in[8];
    float* out = (float*)d_out;

    char* ws = (char*)d_ws;
    size_t off = 0;
    auto alloc = [&](size_t bytes) -> char* {
        char* p = ws + off;
        off += (bytes + 255) & ~(size_t)255;
        return p;
    };
    unsigned short* Xb  = (unsigned short*)alloc((size_t)BS * Dm * 2);
    unsigned short* Wqb = (unsigned short*)alloc((size_t)Dm * Dm * 2);
    unsigned short* Wkb = (unsigned short*)alloc((size_t)Dm * Dm * 2);
    unsigned short* Wvb = (unsigned short*)alloc((size_t)Dm * Dm * 2);
    unsigned short* Wob = (unsigned short*)alloc((size_t)Dm * Dm * 2);
    unsigned short* Qb  = (unsigned short*)alloc((size_t)BS * Dm * 2);
    unsigned short* Kb  = (unsigned short*)alloc((size_t)BS * Dm * 2);
    unsigned short* Vt  = (unsigned short*)alloc((size_t)BS * Dm * 2);
    unsigned short* Ctx = (unsigned short*)alloc((size_t)BS * Dm * 2);
    float*          Yf  = (float*)alloc((size_t)BS * Dm * 4);

    // f32 -> bf16 conversions (X + 4 weights)
    conv_bf16_kernel<<<1024, 256, 0, stream>>>(X,  Xb,  BS * Dm);
    conv_bf16_kernel<<<1024, 256, 0, stream>>>(wq, Wqb, Dm * Dm);
    conv_bf16_kernel<<<1024, 256, 0, stream>>>(wk, Wkb, Dm * Dm);
    conv_bf16_kernel<<<1024, 256, 0, stream>>>(wv, Wvb, Dm * Dm);
    conv_bf16_kernel<<<1024, 256, 0, stream>>>(wo, Wob, Dm * Dm);

    // QKV + rotary: 3 * 128 * 16 = 6144 waves, 8 waves/block
    qkv_kernel<<<768, 256, 0, stream>>>(Xb, Wqb, Wkb, Wvb, sinp, cosp, Qb, Kb, Vt);

    // attention: 32 (b,h) x 16 query-blocks = 512 blocks
    attn_kernel<<<512, 256, 0, stream>>>(Qb, Kb, Vt, Ctx);

    // output projection + residual: 128*16 = 2048 waves
    oproj_kernel<<<256, 256, 0, stream>>>(Ctx, Wob, bo, X, Yf);

    // norm: 4096 rows, one wave each
    norm_kernel<<<512, 256, 0, stream>>>(Yf, scale, out);
}
